// SequenceProjector_70102456205882
// MI455X (gfx1250) — matrix operand order repaired and checked by reading, not compiled, not run on hardware
//
#include <hip/hip_runtime.h>

#define Bsz 2048
#define Mm 4
#define Dd 1280
#define Hh 1024
#define Oo 512

typedef __attribute__((ext_vector_type(16))) __bf16 v16bf;
typedef __attribute__((ext_vector_type(8)))  float  v8f;
typedef __attribute__((ext_vector_type(8)))  unsigned short ushort8;

__device__ __forceinline__ unsigned short f2bf(float f) {
  unsigned u = __float_as_uint(f);
  u += 0x7FFFu + ((u >> 16) & 1u);   // round-to-nearest-even
  return (unsigned short)(u >> 16);
}

union FragU { v16bf v; ushort8 h[2]; };

// ---------------------------------------------------------------------------
// Generic bf16 WMMA GEMM: out = A(bf16, rows x K, stride lda) @ W(bf16, K x N) + bias
// Block tile 128x128x64, 8 waves (wave32), each wave 64x32 via 4x2 WMMA frags,
// 16 v_wmma per staging phase. Register-staged software pipeline: tile k+1 is
// prefetched to VGPRs during compute, stored to LDS after the read barrier.
// ---------------------------------------------------------------------------
#define BK   64
#define LDSS 72   // padded LDS row stride (halves): 144B rows, 16B-aligned chunks

__global__ __launch_bounds__(256) void k_gemm_bf16(
    const unsigned short* __restrict__ A, long long lda, long long aBS,
    const unsigned short* __restrict__ W, long long ldw, long long wBS,
    const float* __restrict__ bias, long long biasBS,
    float* __restrict__ outF, unsigned short* __restrict__ outH,
    long long ldo, long long oBS, int K, int relu)
{
  __shared__ unsigned short sA[128 * LDSS];
  __shared__ unsigned short sB[128 * LDSS];
  const int tid  = threadIdx.x;
  const int lane = tid & 31;
  const int wave = tid >> 5;
  const int wn   = wave & 3;   // n-sub-tile (32 cols)
  const int wm   = wave >> 2;  // m-sub-tile (64 rows)
  const long long z = blockIdx.z;
  const unsigned short* Az = A + z * aBS;
  const unsigned short* Wz = W + z * wBS;
  const float* biasZ = bias + z * biasBS;
  const int m0 = blockIdx.y * 128;
  const int n0 = blockIdx.x * 128;

  v8f acc[4][2];
#pragma unroll
  for (int i = 0; i < 4; ++i)
#pragma unroll
    for (int j = 0; j < 2; ++j)
#pragma unroll
      for (int r = 0; r < 8; ++r) acc[i][j][r] = 0.0f;

  // ---- prologue: stage tile 0 into LDS ----
#pragma unroll
  for (int it = 0; it < 4; ++it) {
    int v = tid + it * 256;
    int r = v >> 3, c = (v & 7) << 3;
    *(ushort8*)(&sA[r * LDSS + c]) =
        *(const ushort8*)(Az + (long long)(m0 + r) * lda + c);
  }
#pragma unroll
  for (int it = 0; it < 2; ++it) {
    int u = tid + it * 256;
    int kr = (u >> 4) << 1, nc = (u & 15) << 3;
    ushort8 d0 = *(const ushort8*)(Wz + (long long)kr * ldw + (n0 + nc));
    ushort8 d1 = *(const ushort8*)(Wz + (long long)(kr + 1) * ldw + (n0 + nc));
#pragma unroll
    for (int e = 0; e < 8; ++e) {
      unsigned pk = (unsigned)d0[e] | ((unsigned)d1[e] << 16);
      *(unsigned*)(&sB[(nc + e) * LDSS + kr]) = pk;
    }
  }
  __syncthreads();

  // ---- main pipelined K loop ----
  for (int k0 = 0;;) {
    const int k1 = k0 + BK;
    const bool more = (k1 < K);

    // prefetch next tile global -> registers (overlaps with WMMA below)
    ushort8 pa[4], pw[4];
    if (more) {
#pragma unroll
      for (int it = 0; it < 4; ++it) {
        int v = tid + it * 256;
        int r = v >> 3, c = (v & 7) << 3;
        pa[it] = *(const ushort8*)(Az + (long long)(m0 + r) * lda + (k1 + c));
      }
#pragma unroll
      for (int it = 0; it < 2; ++it) {
        int u = tid + it * 256;
        int kr = (u >> 4) << 1, nc = (u & 15) << 3;
        pw[2 * it]     = *(const ushort8*)(Wz + (long long)(k1 + kr) * ldw + (n0 + nc));
        pw[2 * it + 1] = *(const ushort8*)(Wz + (long long)(k1 + kr + 1) * ldw + (n0 + nc));
      }
    }

    // compute on current LDS tile: 2 K-slices x (4x2) WMMA
#pragma unroll
    for (int s = 0; s < 2; ++s) {
      FragU aF[4], bF[2];
      const int arow  = lane & 15;
      const int koffA = s * 32 + ((lane >> 4) << 3);
      const int koffB = s * 32 + ((lane >> 4) << 3);
#pragma unroll
      for (int mi = 0; mi < 4; ++mi) {
        const unsigned short* p = &sA[(wm * 64 + mi * 16 + arow) * LDSS + koffA];
        aF[mi].h[0] = *(const ushort8*)(p);
        aF[mi].h[1] = *(const ushort8*)(p + 16);
      }
#pragma unroll
      for (int ni = 0; ni < 2; ++ni) {
        const unsigned short* p = &sB[(wn * 32 + ni * 16 + arow) * LDSS + koffB];
        bF[ni].h[0] = *(const ushort8*)(p);
        bF[ni].h[1] = *(const ushort8*)(p + 16);
      }
#pragma unroll
      for (int mi = 0; mi < 4; ++mi)
#pragma unroll
        for (int ni = 0; ni < 2; ++ni)
          acc[mi][ni] = __builtin_amdgcn_wmma_f32_16x16x32_bf16(
              false, aF[mi].v, false, bF[ni].v, (short)0, acc[mi][ni], false, false);
    }
    __syncthreads();              // all waves done reading current tile
    if (!more) break;

    // store prefetched tile registers -> LDS
#pragma unroll
    for (int it = 0; it < 4; ++it) {
      int v = tid + it * 256;
      int r = v >> 3, c = (v & 7) << 3;
      *(ushort8*)(&sA[r * LDSS + c]) = pa[it];
    }
#pragma unroll
    for (int it = 0; it < 2; ++it) {
      int u = tid + it * 256;
      int kr = (u >> 4) << 1, nc = (u & 15) << 3;
#pragma unroll
      for (int e = 0; e < 8; ++e) {
        unsigned pk = (unsigned)pw[2 * it][e] | ((unsigned)pw[2 * it + 1][e] << 16);
        *(unsigned*)(&sB[(nc + e) * LDSS + kr]) = pk;
      }
    }
    __syncthreads();
    k0 = k1;
  }

  // ---- epilogue: bias (+ReLU) -> fp32 and/or bf16 stores ----
  const int rsel = (lane >> 4) << 3;
  float*          outFz = outF ? outF + z * oBS : nullptr;
  unsigned short* outHz = outH ? outH + z * oBS : nullptr;
#pragma unroll
  for (int ni = 0; ni < 2; ++ni) {
    const int n  = n0 + wn * 32 + ni * 16 + (lane & 15);
    const float bv = biasZ[n];
#pragma unroll
    for (int mi = 0; mi < 4; ++mi) {
      const int rbase = m0 + wm * 64 + mi * 16 + rsel;
#pragma unroll
      for (int r = 0; r < 8; ++r) {
        float val = acc[mi][ni][r] + bv;
        if (relu) val = fmaxf(val, 0.0f);
        const long long idx = (long long)(rbase + r) * ldo + n;
        if (outFz) outFz[idx] = val;
        if (outHz) outHz[idx] = f2bf(val);
      }
    }
  }
}

// ---------------------------------------------------------------------------
// Elementwise / reduction kernels
// ---------------------------------------------------------------------------
__global__ void k_cast(const float* __restrict__ s, unsigned short* __restrict__ d, int n) {
  int i = blockIdx.x * 256 + threadIdx.x;
  if (i < n) d[i] = f2bf(s[i]);
}

__global__ void k_bn(const float* __restrict__ p, const float* __restrict__ g,
                     const float* __restrict__ be, const float* __restrict__ mn,
                     const float* __restrict__ vr, unsigned short* __restrict__ out, int n) {
  int i = blockIdx.x * 256 + threadIdx.x;
  if (i >= n) return;
  int c = i % (Mm * Dd);
  float x = (p[i] - mn[c]) * rsqrtf(vr[c] + 1e-5f) * g[c] + be[c];
  out[i] = f2bf(x);
}

__global__ __launch_bounds__(256) void k_lnrelu(const float* __restrict__ a,
                                                const float* __restrict__ g,
                                                const float* __restrict__ b,
                                                unsigned short* __restrict__ feats) {
  __shared__ float rs[256], rq[256];
  const int row = blockIdx.x;          // b*4 + m
  const int m = row & 3;
  const int tid = threadIdx.x;
  const float* ar = a + (long long)row * Hh;
  float x[4], s = 0.f, q = 0.f;
#pragma unroll
  for (int t = 0; t < 4; ++t) { x[t] = ar[tid + t * 256]; s += x[t]; q += x[t] * x[t]; }
  rs[tid] = s; rq[tid] = q;
  __syncthreads();
  for (int st = 128; st > 0; st >>= 1) {
    if (tid < st) { rs[tid] += rs[tid + st]; rq[tid] += rq[tid + st]; }
    __syncthreads();
  }
  const float mean = rs[0] * (1.0f / Hh);
  const float var  = rq[0] * (1.0f / Hh) - mean * mean;
  const float inv  = rsqrtf(var + 1e-5f);
  unsigned short* fr = feats + (long long)row * Hh;
#pragma unroll
  for (int t = 0; t < 4; ++t) {
    int k = tid + t * 256;
    float v = (x[t] - mean) * inv * g[m * Hh + k] + b[m * Hh + k];
    fr[k] = f2bf(fmaxf(v, 0.f));
  }
}

__global__ __launch_bounds__(256) void k_l2norm(float* __restrict__ p, int rowlen) {
  __shared__ float rq[256];
  float* r = p + (long long)blockIdx.x * rowlen;
  const int tid = threadIdx.x;
  float q = 0.f;
  for (int k = tid; k < rowlen; k += 256) { float v = r[k]; q += v * v; }
  rq[tid] = q; __syncthreads();
  for (int st = 128; st > 0; st >>= 1) { if (tid < st) rq[tid] += rq[tid + st]; __syncthreads(); }
  const float sc = 1.0f / fmaxf(sqrtf(rq[0]), 1e-12f);
  for (int k = tid; k < rowlen; k += 256) r[k] *= sc;
}

// For one (i,j): dot(o_row, f_i_row) and sum(o_row^2) per batch row
__global__ __launch_bounds__(256) void k_dotss(const float* __restrict__ osl,
                                               const float* __restrict__ f,
                                               float* __restrict__ dotb,
                                               float* __restrict__ ssb, int i, int j) {
  __shared__ float rd[256], rq[256];
  const int b = blockIdx.x, tid = threadIdx.x;
  const float* orow = osl + (long long)b * Hh;
  const float* frow = f + (long long)b * (Mm * Hh) + i * Hh;
  float d = 0.f, q = 0.f;
  for (int k = tid; k < Hh; k += 256) { float v = orow[k]; d += v * frow[k]; q += v * v; }
  rd[tid] = d; rq[tid] = q; __syncthreads();
  for (int st = 128; st > 0; st >>= 1) {
    if (tid < st) { rd[tid] += rd[tid + st]; rq[tid] += rq[tid + st]; }
    __syncthreads();
  }
  if (tid == 0) { dotb[b * 16 + i * 4 + j] = rd[0]; ssb[b * 16 + i * 4 + j] = rq[0]; }
}

// alpha softmax over j + y_tilde combine, one block per (b,i)
__global__ __launch_bounds__(128) void k_alpha(const float* __restrict__ dotb,
                                               const float* __restrict__ ssb,
                                               const float* __restrict__ y,
                                               float* __restrict__ ytilde) {
  __shared__ float al[4];
  const int b = blockIdx.x >> 2, i = blockIdx.x & 3, tid = threadIdx.x;
  if (tid == 0) {
    float lg[4];
    for (int j = 0; j < 4; ++j) {
      float nrm = sqrtf(ssb[b * 16 + i * 4 + j]);
      lg[j] = dotb[b * 16 + i * 4 + j] / fmaxf(nrm, 1e-12f) * 2.0f;  // / ALPHA_T(0.5)
    }
    float mx = fmaxf(fmaxf(lg[0], lg[1]), fmaxf(lg[2], lg[3]));
    float e[4], sm = 0.f;
    for (int j = 0; j < 4; ++j) { e[j] = expf(lg[j] - mx); sm += e[j]; }
    for (int j = 0; j < 4; ++j) al[j] = e[j] / sm;
  }
  __syncthreads();
  const float a0 = al[0], a1 = al[1], a2 = al[2], a3 = al[3];
  const float* yb = y + (long long)b * (16 * Oo) + (long long)i * 4 * Oo;
  float* yt = ytilde + (long long)b * (Mm * Oo) + i * Oo;
#pragma unroll
  for (int t = 0; t < 4; ++t) {
    int o = tid + t * 128;
    yt[o] = a0 * yb[o] + a1 * yb[Oo + o] + a2 * yb[2 * Oo + o] + a3 * yb[3 * Oo + o];
  }
}

// beta softmax + top-2 straight-through + final combine, one block per b
__global__ __launch_bounds__(128) void k_beta_final(const float* __restrict__ f,
                                                    const float* __restrict__ q,
                                                    const float* __restrict__ ytilde,
                                                    float* __restrict__ yhat,
                                                    float* __restrict__ maskOut) {
  __shared__ float red[128];
  __shared__ float wv[4], mk[4];
  const int b = blockIdx.x, tid = threadIdx.x;
  const int m = tid >> 5, lane = tid & 31;
  const float* fr = f + (long long)b * (Mm * Hh) + m * Hh;
  const float* qr = q + (long long)b * Hh;
  float d = 0.f;
  for (int k = lane; k < Hh; k += 32) d += fr[k] * qr[k];
  red[tid] = d; __syncthreads();
  for (int st = 16; st > 0; st >>= 1) { if (lane < st) red[tid] += red[tid + st]; __syncthreads(); }
  if (tid == 0) {
    float lg[4]; for (int mm = 0; mm < 4; ++mm) lg[mm] = red[mm * 32];  // BETA_T == 1
    float mx = fmaxf(fmaxf(lg[0], lg[1]), fmaxf(lg[2], lg[3]));
    float e[4], sm = 0.f;
    for (int mm = 0; mm < 4; ++mm) { e[mm] = expf(lg[mm] - mx); sm += e[mm]; }
    float beta[4]; for (int mm = 0; mm < 4; ++mm) beta[mm] = e[mm] / sm;
    int i1 = 0; for (int mm = 1; mm < 4; ++mm) if (beta[mm] > beta[i1]) i1 = mm;
    int i2 = -1;
    for (int mm = 0; mm < 4; ++mm) { if (mm == i1) continue; if (i2 < 0 || beta[mm] > beta[i2]) i2 = mm; }
    float ws = beta[i1] + beta[i2] + 1e-8f;
    for (int mm = 0; mm < 4; ++mm) {
      bool sel = (mm == i1) || (mm == i2);
      mk[mm] = sel ? 1.0f : 0.0f;
      wv[mm] = sel ? beta[mm] / ws : 0.0f;
    }
  }
  __syncthreads();
  const float* yt = ytilde + (long long)b * (Mm * Oo);
#pragma unroll
  for (int t = 0; t < 4; ++t) {
    int o = tid + t * 128;
    yhat[(long long)b * Oo + o] =
        wv[0] * yt[o] + wv[1] * yt[Oo + o] + wv[2] * yt[2 * Oo + o] + wv[3] * yt[3 * Oo + o];
  }
  if (tid < 4) maskOut[b * 4 + tid] = mk[tid];
}

// ---------------------------------------------------------------------------
extern "C" void kernel_launch(void* const* d_in, const int* in_sizes, int n_in,
                              void* d_out, int out_size, void* d_ws, size_t ws_size,
                              hipStream_t stream) {
  (void)in_sizes; (void)n_in; (void)out_size; (void)ws_size;
  const float* protein = (const float*)d_in[0];
  const float* bn_g = (const float*)d_in[1];
  const float* bn_b = (const float*)d_in[2];
  const float* bn_m = (const float*)d_in[3];
  const float* bn_v = (const float*)d_in[4];
  const float* ad_W = (const float*)d_in[5];
  const float* ad_b = (const float*)d_in[6];
  const float* ln_g = (const float*)d_in[7];
  const float* ln_b = (const float*)d_in[8];
  const float* h_W1 = (const float*)d_in[9];
  const float* h_b1 = (const float*)d_in[10];
  const float* h_W2 = (const float*)d_in[11];
  const float* h_b2 = (const float*)d_in[12];
  const float* WF_w = (const float*)d_in[13];
  const float* WF_b = (const float*)d_in[14];
  const float* WO_w = (const float*)d_in[15];
  const float* WO_b = (const float*)d_in[16];
  const float* WQ_w = (const float*)d_in[17];
  const float* WQ_b = (const float*)d_in[18];

  char* ws = (char*)d_ws;
  size_t off = 0;
  auto alloc = [&](size_t bytes) { size_t r = off; off += (bytes + 255) & ~(size_t)255; return r; };
  unsigned short* xbn    = (unsigned short*)(ws + alloc((size_t)Bsz * 5120 * 2));
  unsigned short* adWb   = (unsigned short*)(ws + alloc((size_t)4 * 1280 * 1024 * 2));
  unsigned short* hW1b   = (unsigned short*)(ws + alloc((size_t)4 * 1024 * 1024 * 2));
  unsigned short* hW2b   = (unsigned short*)(ws + alloc((size_t)4 * 1024 * 512 * 2));
  unsigned short* WFb    = (unsigned short*)(ws + alloc((size_t)1024 * 1024 * 2));
  unsigned short* WOb    = (unsigned short*)(ws + alloc((size_t)1024 * 1024 * 2));
  unsigned short* WQb    = (unsigned short*)(ws + alloc((size_t)4096 * 1024 * 2));
  float*          abuf   = (float*)(ws + alloc((size_t)Bsz * 4096 * 4));
  unsigned short* featsb = (unsigned short*)(ws + alloc((size_t)Bsz * 4096 * 2));
  unsigned short* hb     = (unsigned short*)(ws + alloc((size_t)Bsz * 16384 * 2));
  float*          ybuf   = (float*)(ws + alloc((size_t)Bsz * 8192 * 4));
  float*          fbuf   = (float*)(ws + alloc((size_t)Bsz * 4096 * 4));
  float*          qbuf   = (float*)(ws + alloc((size_t)Bsz * 1024 * 4));
  float*          oslice = (float*)(ws + alloc((size_t)Bsz * 1024 * 4));
  float*          dotb   = (float*)(ws + alloc((size_t)Bsz * 16 * 4));
  float*          ssb    = (float*)(ws + alloc((size_t)Bsz * 16 * 4));
  float*          ytil   = (float*)(ws + alloc((size_t)Bsz * 2048 * 4));

  // 1) weights -> bf16
  auto cast = [&](const float* s, unsigned short* d, int n) {
    k_cast<<<(n + 255) / 256, 256, 0, stream>>>(s, d, n);
  };
  cast(ad_W, adWb, 4 * 1280 * 1024);
  cast(h_W1, hW1b, 4 * 1024 * 1024);
  cast(h_W2, hW2b, 4 * 1024 * 512);
  cast(WF_w, WFb, 1024 * 1024);
  cast(WO_w, WOb, 1024 * 1024);
  cast(WQ_w, WQb, 4096 * 1024);

  // 2) BatchNorm -> bf16
  k_bn<<<(Bsz * 5120) / 256, 256, 0, stream>>>(protein, bn_g, bn_b, bn_m, bn_v, xbn, Bsz * 5120);

  // 3) adapters: a = xbn_m @ ad_W_m + b   (batched over m via grid.z)
  k_gemm_bf16<<<dim3(8, 16, 4), 256, 0, stream>>>(
      xbn, 5120, 1280, adWb, 1024, (long long)1280 * 1024, ad_b, 1024,
      abuf, nullptr, 4096, 1024, 1280, 0);

  // 4) LayerNorm + ReLU -> feats (bf16)
  k_lnrelu<<<Bsz * 4, 256, 0, stream>>>(abuf, ln_g, ln_b, featsb);

  // 5) h_ij = relu(feats_i @ h_W1_j + b1_j), per j batched over i
  for (int j = 0; j < 4; ++j)
    k_gemm_bf16<<<dim3(8, 16, 4), 256, 0, stream>>>(
        featsb, 4096, 1024, hW1b + (size_t)j * 1024 * 1024, 1024, 0, h_b1 + j * 1024, 0,
        nullptr, hb + j * 1024, 16384, 4096, 1024, 1);

  // 6) y_ij = h_ij @ h_W2_j + b2_j
  for (int j = 0; j < 4; ++j)
    k_gemm_bf16<<<dim3(4, 16, 4), 256, 0, stream>>>(
        hb + j * 1024, 16384, 4096, hW2b + (size_t)j * 1024 * 512, 512, 0, h_b2 + j * 512, 0,
        ybuf + j * 512, nullptr, 8192, 2048, 1024, 0);

  // 7) f = l2norm(feats @ WF + b)
  k_gemm_bf16<<<dim3(8, 16, 4), 256, 0, stream>>>(
      featsb, 4096, 1024, WFb, 1024, 0, WF_b, 0, fbuf, nullptr, 4096, 1024, 1024, 0);
  k_l2norm<<<Bsz * 4, 256, 0, stream>>>(fbuf, 1024);

  // 8) q = l2norm(feats_flat @ WQ + b)
  k_gemm_bf16<<<dim3(8, 16, 1), 256, 0, stream>>>(
      featsb, 4096, 0, WQb, 1024, 0, WQ_b, 0, qbuf, nullptr, 1024, 0, 4096, 0);
  k_l2norm<<<Bsz, 256, 0, stream>>>(qbuf, 1024);

  // 9) per (i,j): o slice GEMM then reduce to dot / sumsq (avoids 134MB o tensor)
  for (int i = 0; i < 4; ++i)
    for (int j = 0; j < 4; ++j) {
      k_gemm_bf16<<<dim3(8, 16, 1), 256, 0, stream>>>(
          hb + (i * 4 + j) * 1024, 16384, 0, WOb, 1024, 0, WO_b, 0,
          oslice, nullptr, 1024, 0, 1024, 0);
      k_dotss<<<Bsz, 256, 0, stream>>>(oslice, fbuf, dotb, ssb, i, j);
    }

  // 10) alpha softmax + y_tilde
  k_alpha<<<Bsz * 4, 128, 0, stream>>>(dotb, ssb, ybuf, ytil);

  // 11) beta softmax + top-2 + final combine -> (y_hat, mask)
  float* outF = (float*)d_out;
  k_beta_final<<<Bsz, 128, 0, stream>>>(fbuf, qbuf, ytil, outF, outF + (size_t)Bsz * Oo);
}
